// OptimizedALiBiMultiHeadAttention_35940286333415
// MI455X (gfx1250) — compile-verified, matrix operand order checked by reading, not run on hardware
//
#include <hip/hip_runtime.h>
#include <hip/hip_bf16.h>
#include <math.h>

// ---------------------------------------------------------------------------
// ALiBi multi-head attention for MI455X (gfx1250, wave32, WMMA)
//   x      : [4, 2048, 1024] fp32
//   W_kqv  : [3072, 1024]    fp32   (torch Linear: kqv = x @ W^T, order k,q,v)
//   out    : [4, 2048, 1024] fp32
// Pipeline:
//   K0: convert W to bf16 (one pass, 6.3 MB -> lives in L2 for the GEMM)
//   K1: QKV GEMM, each wave = 16x64 C tile (4 WMMA accumulators, A reused 4x),
//       scatter to per-head K [bh][s][d], Q [bh][s][d], V^T [bh][d][s] (bf16)
//   K2: flash-style attention per (b,h,16-query-tile) wave, online softmax
// ---------------------------------------------------------------------------

typedef __attribute__((ext_vector_type(16))) __bf16 v16bf;
typedef __attribute__((ext_vector_type(8)))  __bf16 v8bf;
typedef __attribute__((ext_vector_type(8)))  float  v8f;

#define B_SZ   4
#define SEQ    2048
#define DMODEL 1024
#define NHEAD  16
#define HDIM   64
#define M_TOT  (B_SZ * SEQ)                           // 8192
#define N_TOT  (3 * DMODEL)                           // 3072
#define NQ_CNT (N_TOT / 64)                           // 48 64-wide n-tiles
#define PER_HS ((size_t)B_SZ * NHEAD * SEQ * HDIM)    // elems of each of qb/kb/vT
#define W_ELEM ((size_t)N_TOT * DMODEL)               // 3.1M elems

// Build a 16-element bf16 A/B fragment from two contiguous 8-element chunks
// (CDNA5 16-bit striping: lanes 0-15 hold K {0..7,16..23}, lanes 16-31 +8).
__device__ __forceinline__ v16bf frag_from_bf16(const __bf16* p0, const __bf16* p1) {
    v8bf lo = *(const v8bf*)p0;
    v8bf hi = *(const v8bf*)p1;
    return __builtin_shufflevector(lo, hi, 0,1,2,3,4,5,6,7,8,9,10,11,12,13,14,15);
}

// Same, converting fp32 source chunks to bf16 in registers (v_cvt_pk_bf16_f32).
__device__ __forceinline__ v16bf frag_from_f32(const float* p0, const float* p1) {
    v8f lo = *(const v8f*)p0;
    v8f hi = *(const v8f*)p1;
    v8bf lob = __builtin_convertvector(lo, v8bf);
    v8bf hib = __builtin_convertvector(hi, v8bf);
    return __builtin_shufflevector(lob, hib, 0,1,2,3,4,5,6,7,8,9,10,11,12,13,14,15);
}

__device__ __forceinline__ unsigned short bf16_bits(float f) {
    __bf16 b = (__bf16)f;
    return __builtin_bit_cast(unsigned short, b);
}

// ---------------------------------------------------------------------------
// Kernel 0: fp32 -> bf16 bulk convert (8 elements / thread, b128 traffic)
// ---------------------------------------------------------------------------
__global__ __launch_bounds__(256)
void cvt_bf16_kernel(const float* __restrict__ src, unsigned short* __restrict__ dst) {
    const size_t i = ((size_t)blockIdx.x * blockDim.x + threadIdx.x) * 8;
    v8f v = *(const v8f*)(src + i);
    *(v8bf*)((__bf16*)dst + i) = __builtin_convertvector(v, v8bf);
}

// ---------------------------------------------------------------------------
// Kernel 1: kqv = x @ W^T.  One wave = 16(M) x 64(N) C tile, K-loop of 32.
// N tile of 64 == head_dim, so chunk/head are wave-uniform (scalarized).
// ---------------------------------------------------------------------------
__global__ __launch_bounds__(256)
void qkv_gemm_kernel(const float* __restrict__ x, const unsigned short* __restrict__ wb,
                     unsigned short* __restrict__ qb, unsigned short* __restrict__ kb,
                     unsigned short* __restrict__ vT) {
    const int lane = threadIdx.x & 31;
    const int wave = threadIdx.x >> 5;
    const int tile = blockIdx.x * 8 + wave;       // 0 .. 24575
    const int mt  = tile / NQ_CNT;                // 0 .. 511
    const int ntq = tile - mt * NQ_CNT;           // 0 .. 47  (64-wide n tile)
    const int ln16 = lane & 15;
    const int grp  = (lane >> 4) & 1;
    const int klo  = grp ? 8 : 0;

    const float*  arow = x + (size_t)(mt * 16 + ln16) * DMODEL;
    const __bf16* wrow = (const __bf16*)wb + (size_t)(ntq * 64 + ln16) * DMODEL;

    v8f c0 = {}, c1 = {}, c2 = {}, c3 = {};
#pragma unroll 2
    for (int kk = 0; kk < DMODEL; kk += 32) {
        const v16bf a = frag_from_f32(arow + kk + klo, arow + kk + klo + 16);
        v16bf bm = frag_from_bf16(wrow + kk + klo, wrow + kk + klo + 16);
        c0 = __builtin_amdgcn_wmma_f32_16x16x32_bf16(false, a, false, bm, (short)0, c0, false, false);
        bm = frag_from_bf16(wrow + 16 * DMODEL + kk + klo, wrow + 16 * DMODEL + kk + klo + 16);
        c1 = __builtin_amdgcn_wmma_f32_16x16x32_bf16(false, a, false, bm, (short)0, c1, false, false);
        bm = frag_from_bf16(wrow + 32 * DMODEL + kk + klo, wrow + 32 * DMODEL + kk + klo + 16);
        c2 = __builtin_amdgcn_wmma_f32_16x16x32_bf16(false, a, false, bm, (short)0, c2, false, false);
        bm = frag_from_bf16(wrow + 48 * DMODEL + kk + klo, wrow + 48 * DMODEL + kk + klo + 16);
        c3 = __builtin_amdgcn_wmma_f32_16x16x32_bf16(false, a, false, bm, (short)0, c3, false, false);
    }

    // ---- scatter: n-range [ntq*64, ntq*64+64) == one (chunk, head) -------
    const int chunk = __builtin_amdgcn_readfirstlane(ntq >> 4);   // 0=k,1=q,2=v
    const int h     = __builtin_amdgcn_readfirstlane(ntq & 15);
    const int mbase = mt * 16 + grp * 8;
    const int b     = mbase >> 11;
    const int sbase = mbase & (SEQ - 1);
    const size_t bh = (size_t)(b * NHEAD + h);

    if (chunk == 2) {
        // V^T [bh][d][s]: rows r are contiguous in s -> one b128 store / frag
        v8bf p;
        p = __builtin_convertvector(c0, v8bf);
        *(v8bf*)((__bf16*)vT + (bh * HDIM + ln16)      * SEQ + sbase) = p;
        p = __builtin_convertvector(c1, v8bf);
        *(v8bf*)((__bf16*)vT + (bh * HDIM + 16 + ln16) * SEQ + sbase) = p;
        p = __builtin_convertvector(c2, v8bf);
        *(v8bf*)((__bf16*)vT + (bh * HDIM + 32 + ln16) * SEQ + sbase) = p;
        p = __builtin_convertvector(c3, v8bf);
        *(v8bf*)((__bf16*)vT + (bh * HDIM + 48 + ln16) * SEQ + sbase) = p;
    } else {
        unsigned short* dst = (chunk == 0) ? kb : qb;
        unsigned short* drow = dst + (bh * SEQ + sbase) * HDIM + ln16;
#pragma unroll
        for (int r = 0; r < 8; ++r) {
            drow[(size_t)r * HDIM]      = bf16_bits(c0[r]);
            drow[(size_t)r * HDIM + 16] = bf16_bits(c1[r]);
            drow[(size_t)r * HDIM + 32] = bf16_bits(c2[r]);
            drow[(size_t)r * HDIM + 48] = bf16_bits(c3[r]);
        }
    }
}

// ---------------------------------------------------------------------------
// Kernel 2: flash attention with ALiBi bias.  One wave per (bh, 16-row q tile).
// ---------------------------------------------------------------------------
__global__ __launch_bounds__(256)
void alibi_attn_kernel(const unsigned short* __restrict__ qb,
                       const unsigned short* __restrict__ kb,
                       const unsigned short* __restrict__ vT,
                       float* __restrict__ out) {
    __shared__ unsigned short psh[8][16 * 32];    // per-wave P re-striping buffer

    const int lane = threadIdx.x & 31;
    const int wave = threadIdx.x >> 5;
    const int task = blockIdx.x * 8 + wave;       // 0 .. 8191
    const int bh = task >> 7;                     // 0 .. 63
    const int qt = task & 127;
    const int b  = bh >> 4;
    const int h  = bh & 15;
    const int ln16 = lane & 15;
    const int grp  = (lane >> 4) & 1;
    const int klo  = grp ? 8 : 0;

    const float slope     = exp2f(-0.5f * (float)(h + 1));  // (2^8)^(-(h+1)/16)
    const float inv_scale = 0.03125f;                       // 1 / sqrt(1024)

    const __bf16* qrow = (const __bf16*)qb + (size_t)(bh * SEQ + qt * 16 + ln16) * HDIM;
    const v16bf aq0 = frag_from_bf16(qrow + klo,      qrow + klo + 16);
    const v16bf aq1 = frag_from_bf16(qrow + 32 + klo, qrow + 48 + klo);

    const __bf16* kbase = (const __bf16*)kb + (size_t)bh * SEQ * HDIM;
    const __bf16* vbase = (const __bf16*)vT + (size_t)bh * HDIM * SEQ;
    unsigned short* pw = psh[wave];

    float m_run[8], l_run[8];
    v8f o0 = {}, o1 = {}, o2 = {}, o3 = {};
#pragma unroll
    for (int r = 0; r < 8; ++r) { m_run[r] = -3.0e38f; l_run[r] = 0.0f; }

    for (int jb = 0; jb < SEQ; jb += 32) {
        // ---- scores for keys [jb, jb+32): two 16x16 C tiles --------------
        const __bf16* krow0 = kbase + (size_t)(jb + ln16) * HDIM;
        const __bf16* krow1 = kbase + (size_t)(jb + 16 + ln16) * HDIM;
        v8f s0 = {}, s1 = {};
        {
            v16bf bk = frag_from_bf16(krow0 + klo, krow0 + klo + 16);
            s0 = __builtin_amdgcn_wmma_f32_16x16x32_bf16(false, aq0, false, bk, (short)0, s0, false, false);
            bk = frag_from_bf16(krow0 + 32 + klo, krow0 + 48 + klo);
            s0 = __builtin_amdgcn_wmma_f32_16x16x32_bf16(false, aq1, false, bk, (short)0, s0, false, false);
            bk = frag_from_bf16(krow1 + klo, krow1 + klo + 16);
            s1 = __builtin_amdgcn_wmma_f32_16x16x32_bf16(false, aq0, false, bk, (short)0, s1, false, false);
            bk = frag_from_bf16(krow1 + 32 + klo, krow1 + 48 + klo);
            s1 = __builtin_amdgcn_wmma_f32_16x16x32_bf16(false, aq1, false, bk, (short)0, s1, false, false);
        }

        if (jb + 32 < SEQ) {  // prefetch next K/V block (global_prefetch_b8)
            __builtin_prefetch(kbase + (size_t)(jb + 32 + ln16) * HDIM, 0, 1);
            __builtin_prefetch(vbase + (size_t)ln16 * SEQ + jb + 32, 0, 1);
        }

        // ---- ALiBi bias + online softmax (row reductions over 16 lanes) --
        const int j0 = jb + ln16;
        const int j1 = j0 + 16;
#pragma unroll
        for (int r = 0; r < 8; ++r) {
            const int  i  = qt * 16 + r + grp * 8;
            float v0 = s0[r] * inv_scale + slope * (float)(j0 - i);
            float v1 = s1[r] * inv_scale + slope * (float)(j1 - i);

            float tmax = fmaxf(v0, v1);
            for (int mk = 1; mk < 16; mk <<= 1)
                tmax = fmaxf(tmax, __shfl_xor(tmax, mk, 32));
            const float mnew  = fmaxf(m_run[r], tmax);
            const float alpha = __expf(m_run[r] - mnew);
            const float p0 = __expf(v0 - mnew);
            const float p1 = __expf(v1 - mnew);
            float psum = p0 + p1;
            for (int mk = 1; mk < 16; mk <<= 1)
                psum += __shfl_xor(psum, mk, 32);
            l_run[r] = l_run[r] * alpha + psum;
            m_run[r] = mnew;
            o0[r] *= alpha; o1[r] *= alpha; o2[r] *= alpha; o3[r] *= alpha;

            const int row = r + grp * 8;
            pw[row * 32 + ln16]      = bf16_bits(p0);
            pw[row * 32 + ln16 + 16] = bf16_bits(p1);
        }

        // ---- restripe P (C layout -> A layout) through LDS ---------------
        const __bf16* prow = (const __bf16*)pw + ln16 * 32;
        const v16bf ap = frag_from_bf16(prow + klo, prow + klo + 16);

        // ---- O += P @ V  (V^T is dim-major: contiguous keys per lane) ----
        const __bf16* v0p = vbase + (size_t)(ln16) * SEQ + jb;
        const __bf16* v1p = vbase + (size_t)(16 + ln16) * SEQ + jb;
        const __bf16* v2p = vbase + (size_t)(32 + ln16) * SEQ + jb;
        const __bf16* v3p = vbase + (size_t)(48 + ln16) * SEQ + jb;
        v16bf bv = frag_from_bf16(v0p + klo, v0p + klo + 16);
        o0 = __builtin_amdgcn_wmma_f32_16x16x32_bf16(false, ap, false, bv, (short)0, o0, false, false);
        bv = frag_from_bf16(v1p + klo, v1p + klo + 16);
        o1 = __builtin_amdgcn_wmma_f32_16x16x32_bf16(false, ap, false, bv, (short)0, o1, false, false);
        bv = frag_from_bf16(v2p + klo, v2p + klo + 16);
        o2 = __builtin_amdgcn_wmma_f32_16x16x32_bf16(false, ap, false, bv, (short)0, o2, false, false);
        bv = frag_from_bf16(v3p + klo, v3p + klo + 16);
        o3 = __builtin_amdgcn_wmma_f32_16x16x32_bf16(false, ap, false, bv, (short)0, o3, false, false);
    }

    // ---- normalize and store ---------------------------------------------
    float* obase = out + ((size_t)b * SEQ) * DMODEL + (size_t)h * HDIM + ln16;
#pragma unroll
    for (int r = 0; r < 8; ++r) {
        const int s = qt * 16 + r + grp * 8;
        const float inv_l = 1.0f / l_run[r];
        float* po = obase + (size_t)s * DMODEL;
        po[0]  = o0[r] * inv_l;
        po[16] = o1[r] * inv_l;
        po[32] = o2[r] * inv_l;
        po[48] = o3[r] * inv_l;
    }
}

// ---------------------------------------------------------------------------
extern "C" void kernel_launch(void* const* d_in, const int* in_sizes, int n_in,
                              void* d_out, int out_size, void* d_ws, size_t ws_size,
                              hipStream_t stream) {
    const float* x = (const float*)d_in[0];       // [4,2048,1024]
    const float* w = (const float*)d_in[1];       // [3072,1024]
    float* out     = (float*)d_out;               // [4,2048,1024]

    unsigned short* qb = (unsigned short*)d_ws;
    unsigned short* kb = qb + PER_HS;
    unsigned short* vT = kb + PER_HS;
    unsigned short* wb = vT + PER_HS;             // + 6.3 MB bf16 weights

    // K0: W fp32 -> bf16 (3.1M elems, 8/thread)
    cvt_bf16_kernel<<<(int)(W_ELEM / 8 / 256), 256, 0, stream>>>(w, wb);

    // K1: 512 m-tiles * 48 n64-tiles = 24576 waves, 8 waves/block
    qkv_gemm_kernel<<<(M_TOT / 16) * NQ_CNT / 8, 256, 0, stream>>>(x, wb, qb, kb, vT);

    // K2: 64 heads * 128 q-tiles = 8192 waves, 8 waves/block
    alibi_attn_kernel<<<(B_SZ * NHEAD * (SEQ / 16)) / 8, 256, 0, stream>>>(qb, kb, vT, out);
}
